// HH3DVFE_28707561406507
// MI455X (gfx1250) — matrix operand order repaired and checked by reading, not compiled, not run on hardware
//
#include <hip/hip_runtime.h>
#include <hip/hip_bf16.h>

// ---------------------------------------------------------------------------
// MVF voxel-feature-encoder pipeline for gfx1250 (MI455X), bf16 WMMA path.
// Weights are pre-packed into WMMA B-fragment layout so GEMM/conv inner loops
// are pure global_load_b128 + v_wmma (no branches, no in-loop converts).
// ---------------------------------------------------------------------------

typedef __attribute__((ext_vector_type(16))) __bf16 v16bf;
typedef __attribute__((ext_vector_type(8)))  float  v8f;

#define FD       64
#define GCX      512            // cyl phi bins (image H)
#define GCY      48             // cyl z bins   (image W)
#define CYL_SXY  (GCX * GCY)    // 24576
#define CYL_SY   GCY            // 48
#define BEV_SXY  (220 * 250)    // 55000
#define BEV_SY   250
#define BN_EPS   0.001f
#define CYL_VS0  0.01227184630f // 2*pi/512
#define CYL_VS1  0.125f         // 6/48

__device__ __forceinline__ __bf16 f2bf(float f) {
  unsigned u = __builtin_bit_cast(unsigned, f);
  unsigned r = (u + 0x7FFFu + ((u >> 16) & 1u)) >> 16;   // round-nearest-even
  return __builtin_bit_cast(__bf16, (unsigned short)r);
}
__device__ __forceinline__ __bf16 bf0() {
  return __builtin_bit_cast(__bf16, (unsigned short)0);
}

// ---------------------------------------------------------------------------
// 1) per-voxel sums (for cluster means)
// ---------------------------------------------------------------------------
__global__ void k_point_stats(const float* __restrict__ points,
                              const float* __restrict__ pcyl,
                              const int* __restrict__ binv,
                              const int* __restrict__ cinv,
                              float* __restrict__ bevsum,
                              float* __restrict__ cylsum, int N) {
  int p = blockIdx.x * blockDim.x + threadIdx.x;
  if (p >= N) return;
  int bv = binv[p], cv = cinv[p];
  atomicAdd(&bevsum[(size_t)bv * 4 + 0], points[(size_t)p * 5 + 1]);
  atomicAdd(&bevsum[(size_t)bv * 4 + 1], points[(size_t)p * 5 + 2]);
  atomicAdd(&bevsum[(size_t)bv * 4 + 2], points[(size_t)p * 5 + 3]);
  atomicAdd(&bevsum[(size_t)bv * 4 + 3], 1.0f);
  atomicAdd(&cylsum[(size_t)cv * 4 + 0], pcyl[(size_t)p * 3 + 0]);
  atomicAdd(&cylsum[(size_t)cv * 4 + 1], pcyl[(size_t)p * 3 + 1]);
  atomicAdd(&cylsum[(size_t)cv * 4 + 2], pcyl[(size_t)p * 3 + 2]);
  atomicAdd(&cylsum[(size_t)cv * 4 + 3], 1.0f);
}

// ---------------------------------------------------------------------------
// 2) build 16 mvf features per point, store bf16 A1 [N,32] (K padded to 32)
// ---------------------------------------------------------------------------
__global__ void k_build_mvfin(const float* __restrict__ points,
                              const float* __restrict__ pcyl,
                              const float* __restrict__ cyl_idxs,
                              const float* __restrict__ bev_idxs,
                              const int* __restrict__ binv,
                              const int* __restrict__ cinv,
                              const float* __restrict__ bevsum,
                              const float* __restrict__ cylsum,
                              __bf16* __restrict__ A1, int N) {
  int p = blockIdx.x * blockDim.x + threadIdx.x;
  if (p >= N) return;
  float x = points[(size_t)p * 5 + 1], y = points[(size_t)p * 5 + 2];
  float z = points[(size_t)p * 5 + 3], it = points[(size_t)p * 5 + 4];
  float ph = pcyl[(size_t)p * 3 + 0], zc = pcyl[(size_t)p * 3 + 1];
  float rh = pcyl[(size_t)p * 3 + 2];
  int bv = binv[p], cv = cinv[p];
  float bcn = bevsum[(size_t)bv * 4 + 3];
  float bmx = bevsum[(size_t)bv * 4 + 0] / bcn;
  float bmy = bevsum[(size_t)bv * 4 + 1] / bcn;
  float ccn = cylsum[(size_t)cv * 4 + 3];
  float cmp = cylsum[(size_t)cv * 4 + 0] / ccn;
  float cmz = cylsum[(size_t)cv * 4 + 1] / ccn;
  float bix = floorf(bev_idxs[(size_t)p * 2 + 0]);
  float biy = floorf(bev_idxs[(size_t)p * 2 + 1]);
  float cix = floorf(cyl_idxs[(size_t)p * 2 + 0]);
  float ciy = floorf(cyl_idxs[(size_t)p * 2 + 1]);
  float f[16];
  f[0] = x; f[1] = y; f[2] = z; f[3] = ph; f[4] = zc; f[5] = rh;
  f[6] = x - ((bix + 0.5f) * 0.32f + 0.0f);
  f[7] = y - ((biy + 0.5f) * 0.32f - 40.0f);
  f[8] = ph - ((cix + 0.5f) * CYL_VS0 - 3.14159265f);
  f[9] = zc - ((ciy + 0.5f) * CYL_VS1 - 2.0f);
  f[10] = x - bmx; f[11] = y - bmy;
  f[12] = ph - cmp; f[13] = zc - cmz;
  f[14] = sqrtf(x * x + y * y + z * z);
  f[15] = it;
  __bf16* d = A1 + (size_t)p * 32;
#pragma unroll
  for (int j = 0; j < 16; ++j) d[j] = f2bf(f[j]);
#pragma unroll
  for (int j = 16; j < 32; ++j) d[j] = bf0();
}

// ---------------------------------------------------------------------------
// 3) pack weight W[Ncols,Kw](f32) -> bf16 B-fragment stream:
//    Bp[((kt*NT + nt)*32 + lane)*16 + v], kk = kt*32 + (lane>=16?8:0) + (v&7) + (v>=8?16:0),
//    value = (kk<Kw) ? W[(nt*16 + lane&15)*Kw + kk] : 0
// ---------------------------------------------------------------------------
__global__ void k_pack_w(const float* __restrict__ W, int Kw, int Kpad,
                         int Ncols, __bf16* __restrict__ Bp) {
  int NT = Ncols >> 4;
  int total = (Kpad >> 5) * NT * 32 * 16;
  int idx = blockIdx.x * blockDim.x + threadIdx.x;
  if (idx >= total) return;
  int v = idx & 15;
  int lane = (idx >> 4) & 31;
  int ntkt = idx >> 9;
  int nt = ntkt % NT;
  int kt = ntkt / NT;
  int kk = kt * 32 + ((lane >> 4) ? 8 : 0) + (v & 7) + ((v >> 3) ? 16 : 0);
  int n = nt * 16 + (lane & 15);
  float val = (kk < Kw) ? W[(size_t)n * Kw + kk] : 0.0f;
  Bp[idx] = f2bf(val);
}

// conv weights wc[64][64][3][3] -> Bp[(((tap*2+kt)*4+nt)*32+lane)*16+v]
__global__ void k_pack_wc(const float* __restrict__ wc,
                          __bf16* __restrict__ Bp) {
  const int total = 9 * 2 * 4 * 32 * 16;
  int idx = blockIdx.x * blockDim.x + threadIdx.x;
  if (idx >= total) return;
  int v = idx & 15;
  int lane = (idx >> 4) & 31;
  int rest = idx >> 9;
  int nt = rest & 3;
  int kt = (rest >> 2) & 1;
  int tap = rest >> 3;
  int kh = tap / 3, kw = tap % 3;
  int ic = kt * 32 + ((lane >> 4) ? 8 : 0) + (v & 7) + ((v >> 3) ? 16 : 0);
  int oc = nt * 16 + (lane & 15);
  Bp[idx] = f2bf(wc[(((size_t)oc * FD + ic) * 3 + kh) * 3 + kw]);
}

// ---------------------------------------------------------------------------
// 4) GEMM: C[M,NT*16] = A[M,Kpad](bf16) @ packed-B. One wave per m-tile,
//    all NT n-tiles per wave (A fragment reused). 4 waves per block.
// ---------------------------------------------------------------------------
template <int NT>
__global__ void k_gemm_bf16(const __bf16* __restrict__ A, int lda,
                            const __bf16* __restrict__ Bp,
                            float* __restrict__ C, int ldc, int Ksteps) {
  int wid = threadIdx.x >> 5;
  int lane = threadIdx.x & 31;
  int half = lane >> 4, l15 = lane & 15;
  int tm = (blockIdx.x * 4 + wid) << 4;
  v8f acc[NT];
#pragma unroll
  for (int nt = 0; nt < NT; ++nt) acc[nt] = v8f{};
  const __bf16* Arow = A + (size_t)(tm + l15) * lda + (half ? 8 : 0);
  for (int kt = 0; kt < Ksteps; ++kt) {
    v16bf a;
    const __bf16* ap = Arow + kt * 32;
#pragma unroll
    for (int v = 0; v < 8; ++v) { a[v] = ap[v]; a[8 + v] = ap[16 + v]; }
    const __bf16* bpk = Bp + ((size_t)(kt * NT) * 32 + lane) * 16;
#pragma unroll
    for (int nt = 0; nt < NT; ++nt) {
      v16bf b;
      const __bf16* bp = bpk + (size_t)nt * 32 * 16;
#pragma unroll
      for (int v = 0; v < 16; ++v) b[v] = bp[v];
      acc[nt] = __builtin_amdgcn_wmma_f32_16x16x32_bf16(
          false, a, false, b, (short)0, acc[nt], false, false);
    }
  }
  float* Crow = C + (size_t)tm * ldc + l15;
#pragma unroll
  for (int nt = 0; nt < NT; ++nt)
#pragma unroll
    for (int r = 0; r < 8; ++r)
      Crow[(size_t)(r + half * 8) * ldc + nt * 16] = acc[nt][r];
}

// ---------------------------------------------------------------------------
// 5) per-column sum/sumsq (coalesced: 256/ld rows per block-iteration)
// ---------------------------------------------------------------------------
__global__ void k_colstats(const float* __restrict__ X, int M, int ld,
                           int ldshift, float* __restrict__ sums,
                           float* __restrict__ sumsq) {
  __shared__ float sh[256], shq[256];
  int t = threadIdx.x;
  int col = t & (ld - 1);
  int rsub = t >> ldshift;
  int rpb = 256 >> ldshift;
  float s = 0.f, q = 0.f;
  for (size_t r = (size_t)blockIdx.x * rpb + rsub; r < (size_t)M;
       r += (size_t)gridDim.x * rpb) {
    float v = X[(r << ldshift) + col];
    s += v; q += v * v;
  }
  sh[t] = s; shq[t] = q;
  __syncthreads();
  for (int st = 128; st >= ld; st >>= 1) {
    if (t < st) { sh[t] += sh[t + st]; shq[t] += shq[t + st]; }
    __syncthreads();
  }
  if (t < ld) { atomicAdd(&sums[col], sh[t]); atomicAdd(&sumsq[col], shq[t]); }
}

// ---------------------------------------------------------------------------
// 6) apply BN + ReLU; optional f32 and/or bf16 destinations
// ---------------------------------------------------------------------------
__global__ void k_bnrelu(const float* __restrict__ X,
                         float* __restrict__ Yf,
                         __bf16* __restrict__ Yb, int ldb, int coloffb,
                         const float* __restrict__ sums,
                         const float* __restrict__ sumsq,
                         const float* __restrict__ g,
                         const float* __restrict__ bb,
                         int M, int ldshift) {
  int ld = 1 << ldshift;
  size_t total = (size_t)M << ldshift;
  float invM = 1.0f / (float)M;
  for (size_t i = (size_t)blockIdx.x * blockDim.x + threadIdx.x; i < total;
       i += (size_t)gridDim.x * blockDim.x) {
    int col = (int)(i & (ld - 1));
    size_t row = i >> ldshift;
    float mean = sums[col] * invM;
    float var  = sumsq[col] * invM - mean * mean;
    float y = (X[i] - mean) * rsqrtf(var + BN_EPS) * g[col] + bb[col];
    y = fmaxf(y, 0.0f);
    if (Yf) Yf[i] = y;
    if (Yb) Yb[row * ldb + coloffb + col] = f2bf(y);
  }
}

// ---------------------------------------------------------------------------
// 7) scatter pt_fea_cyl (h2 cols 0..63) -> dense image via atomic max
// ---------------------------------------------------------------------------
__global__ void k_scatter_cylmax(const float* __restrict__ h2,
                                 const int* __restrict__ cinv,
                                 const int* __restrict__ ccoords,
                                 float* __restrict__ dense, int N) {
  int p = blockIdx.x, f = threadIdx.x;
  if (p >= N) return;
  int c = ccoords[cinv[p]];
  int b  = c / CYL_SXY;
  int rm = c % CYL_SXY;
  int cx = rm / CYL_SY;     // phi bin
  int cy = rm % CYL_SY;     // z bin
  float v = h2[(size_t)p * 128 + f];
  atomicMax((int*)&dense[(((size_t)b * GCX + cx) * GCY + cy) * FD + f],
            __float_as_int(v));
}

__global__ void k_f32_to_bf16(const float* __restrict__ src,
                              __bf16* __restrict__ dst, size_t n) {
  for (size_t i = (size_t)blockIdx.x * blockDim.x + threadIdx.x; i < n;
       i += (size_t)gridDim.x * blockDim.x)
    dst[i] = f2bf(src[i]);
}

// h2 pointwise half (cols 64..127) -> A3 cols 0..63
__global__ void k_pointwise_to_A3(const float* __restrict__ h2,
                                  __bf16* __restrict__ A3, int N) {
  size_t total = (size_t)N * FD;
  for (size_t i = (size_t)blockIdx.x * blockDim.x + threadIdx.x; i < total;
       i += (size_t)gridDim.x * blockDim.x) {
    size_t r = i >> 6; int c = (int)(i & 63);
    A3[r * 128 + c] = f2bf(h2[r * 128 + 64 + c]);
  }
}

// ---------------------------------------------------------------------------
// 8) 3x3 conv (64->64, SAME) implicit-GEMM WMMA. NHWC bf16 [B,512,48,64].
//    One wave per 16-pixel tile, all 4 oc-tiles. 4 waves per block.
// ---------------------------------------------------------------------------
__global__ void k_conv_wmma(const __bf16* __restrict__ in,
                            const __bf16* __restrict__ Bp,
                            float* __restrict__ out) {
  int wid = threadIdx.x >> 5;
  int lane = threadIdx.x & 31;
  int half = lane >> 4, l15 = lane & 15;
  int pt = blockIdx.x * 4 + wid;
  int wt = pt % (GCY / 16);
  int bh = pt / (GCY / 16);
  int h = bh % GCX;
  int b = bh / GCX;
  int wpix = (wt << 4) + l15;
  v8f acc[4];
#pragma unroll
  for (int nt = 0; nt < 4; ++nt) acc[nt] = v8f{};
  for (int kh = 0; kh < 3; ++kh) {
    int hh = h + kh - 1;
    bool hok = (hh >= 0) && (hh < GCX);
    for (int kw = 0; kw < 3; ++kw) {
      int wp = wpix + kw - 1;
      bool ok = hok && (wp >= 0) && (wp < GCY);
      const __bf16* src =
          in + (((size_t)b * GCX + hh) * GCY + wp) * FD + (half ? 8 : 0);
      int tap = kh * 3 + kw;
#pragma unroll
      for (int kt = 0; kt < 2; ++kt) {
        v16bf a;
        if (ok) {
          const __bf16* ap = src + kt * 32;
#pragma unroll
          for (int v = 0; v < 8; ++v) { a[v] = ap[v]; a[8 + v] = ap[16 + v]; }
        } else {
#pragma unroll
          for (int v = 0; v < 16; ++v) a[v] = bf0();
        }
        const __bf16* bpk = Bp + ((size_t)((tap * 2 + kt) * 4) * 32 + lane) * 16;
#pragma unroll
        for (int nt = 0; nt < 4; ++nt) {
          v16bf bf;
          const __bf16* bp = bpk + (size_t)nt * 32 * 16;
#pragma unroll
          for (int v = 0; v < 16; ++v) bf[v] = bp[v];
          acc[nt] = __builtin_amdgcn_wmma_f32_16x16x32_bf16(
              false, a, false, bf, (short)0, acc[nt], false, false);
        }
      }
    }
  }
  float* dst = out + (((size_t)b * GCX + h) * GCY + (wt << 4)) * FD + l15;
#pragma unroll
  for (int nt = 0; nt < 4; ++nt)
#pragma unroll
    for (int r = 0; r < 8; ++r)
      dst[(size_t)(r + half * 8) * FD + nt * 16] = acc[nt][r];
}

// ---------------------------------------------------------------------------
// 9) bilinear gather point_cyl -> A3 cols 64..127 (bf16)
// ---------------------------------------------------------------------------
__global__ void k_bilinear(const float* __restrict__ img,
                           const float* __restrict__ cyl_idxs,
                           const float* __restrict__ points,
                           __bf16* __restrict__ A3, int N) {
  int p = blockIdx.x, f = threadIdx.x;
  if (p >= N) return;
  float yq = cyl_idxs[(size_t)p * 2 + 0];   // phi coord (0..512)
  float xq = cyl_idxs[(size_t)p * 2 + 1];   // z coord   (0..48)
  int b = (int)points[(size_t)p * 5 + 0];
  int xf = (int)floorf(xq), yf = (int)floorf(yq);
  int x0 = min(max(xf, 0), GCY - 1), x1 = min(max(xf + 1, 0), GCY - 1);
  int y0 = min(max(yf, 0), GCX - 1), y1 = min(max(yf + 1, 0), GCX - 1);
  float x0f = (float)x0, x1f = (float)x1, y0f = (float)y0, y1f = (float)y1;
  float wa = (x1f - xq) * (y1f - yq);
  float wb = (x1f - xq) * (yq - y0f);
  float wcw = (xq - x0f) * (y1f - yq);
  float wd = (xq - x0f) * (yq - y0f);
  size_t base = (size_t)b * GCX;
  float Ia = img[(((base + y0) * GCY) + x0) * FD + f];
  float Ib = img[(((base + y1) * GCY) + x0) * FD + f];
  float Ic = img[(((base + y0) * GCY) + x1) * FD + f];
  float Id = img[(((base + y1) * GCY) + x1) * FD + f];
  A3[(size_t)p * 128 + 64 + f] = f2bf(Ia * wa + Ib * wb + Ic * wcw + Id * wd);
}

// ---------------------------------------------------------------------------
// 10) bev segment max + voxel coords
// ---------------------------------------------------------------------------
__global__ void k_bevmax(const float* __restrict__ mvf,
                         const int* __restrict__ binv,
                         float* __restrict__ bevmax, int N) {
  int p = blockIdx.x, f = threadIdx.x;
  if (p >= N) return;
  atomicMax((int*)&bevmax[(size_t)binv[p] * FD + f],
            __float_as_int(mvf[(size_t)p * FD + f]));
}

__global__ void k_voxcoords(const int* __restrict__ bcoords,
                            float* __restrict__ out, int Nb) {
  int i = blockIdx.x * blockDim.x + threadIdx.x;
  if (i >= Nb) return;
  int c = bcoords[i];
  int vb = c / BEV_SXY;
  int vx = (c % BEV_SXY) / BEV_SY;
  int vy = c % BEV_SY;
  out[(size_t)i * 4 + 0] = (float)vb;
  out[(size_t)i * 4 + 1] = 0.0f;
  out[(size_t)i * 4 + 2] = (float)vy;
  out[(size_t)i * 4 + 3] = (float)vx;
}

// ---------------------------------------------------------------------------
extern "C" void kernel_launch(void* const* d_in, const int* in_sizes, int n_in,
                              void* d_out, int out_size, void* d_ws,
                              size_t ws_size, hipStream_t stream) {
  const float* points   = (const float*)d_in[0];
  const float* pcyl     = (const float*)d_in[1];
  const float* cyl_idxs = (const float*)d_in[2];
  const float* bev_idxs = (const float*)d_in[3];
  const float* w1 = (const float*)d_in[4];
  const float* g1 = (const float*)d_in[5];
  const float* b1 = (const float*)d_in[6];
  const float* w2 = (const float*)d_in[7];
  const float* g2 = (const float*)d_in[8];
  const float* b2 = (const float*)d_in[9];
  const float* wc = (const float*)d_in[10];
  const float* gc = (const float*)d_in[11];
  const float* bc = (const float*)d_in[12];
  const float* w3 = (const float*)d_in[13];
  const float* g3 = (const float*)d_in[14];
  const float* b3 = (const float*)d_in[15];
  const int* binv    = (const int*)d_in[16];
  const int* bcoords = (const int*)d_in[17];
  const int* cinv    = (const int*)d_in[18];
  const int* ccoords = (const int*)d_in[19];

  const int N  = in_sizes[0] / 5;     // 200000, divisible by 64
  const int Nb = in_sizes[17];
  const int Nc = in_sizes[19];
  const int B  = 4;                   // batch_size fixed by harness setup
  const int P  = B * GCX * GCY;       // dense image pixels

  // ---- workspace layout ----
  char* w = (char*)d_ws;
  size_t off = 0;
  float* bevsum = (float*)(w + off); off += (size_t)Nb * 4 * sizeof(float);
  float* cylsum = (float*)(w + off); off += (size_t)Nc * 4 * sizeof(float);
  float* st1 = (float*)(w + off); off += 128 * sizeof(float);
  float* st2 = (float*)(w + off); off += 256 * sizeof(float);
  float* stC = (float*)(w + off); off += 128 * sizeof(float);
  float* st3 = (float*)(w + off); off += 128 * sizeof(float);
  float* dense = (float*)(w + off); off += (size_t)P * FD * sizeof(float);
  size_t zbytes = off;                // <- everything above needs zero-init
  __bf16* A1 = (__bf16*)(w + off); off += (size_t)N * 32 * sizeof(__bf16);
  float* h1  = (float*)(w + off);  off += (size_t)N * 64 * sizeof(float);
  __bf16* A2 = (__bf16*)(w + off); off += (size_t)N * 64 * sizeof(__bf16);
  float* h2  = (float*)(w + off);  off += (size_t)N * 128 * sizeof(float);
  __bf16* dbf = (__bf16*)(w + off); off += (size_t)P * FD * sizeof(__bf16);
  float* cvo = (float*)(w + off);  off += (size_t)P * FD * sizeof(float);
  __bf16* A3 = (__bf16*)(w + off); off += (size_t)N * 128 * sizeof(__bf16);
  float* h3  = (float*)(w + off);  off += (size_t)N * 64 * sizeof(float);
  __bf16* Bp1 = (__bf16*)(w + off); off += (size_t)1 * 4 * 512 * sizeof(__bf16);
  __bf16* Bp2 = (__bf16*)(w + off); off += (size_t)2 * 8 * 512 * sizeof(__bf16);
  __bf16* Bp3 = (__bf16*)(w + off); off += (size_t)4 * 4 * 512 * sizeof(__bf16);
  __bf16* BpC = (__bf16*)(w + off); off += (size_t)9 * 2 * 4 * 512 * sizeof(__bf16);

  float* out_mvf = (float*)d_out;               // [N,64]
  float* out_bev = out_mvf + (size_t)N * FD;    // [Nb,64]
  float* out_vc  = out_bev + (size_t)Nb * FD;   // [Nb,4]

  hipMemsetAsync(d_ws, 0, zbytes, stream);
  hipMemsetAsync(d_out, 0, (size_t)out_size * sizeof(float), stream);

  const int tpb = 256;
  int nb256 = (N + tpb - 1) / tpb;

  // weight packing (tiny)
  k_pack_w<<<(1 * 4 * 512 + 255) / 256, 256, 0, stream>>>(w1, 16, 32, 64, Bp1);
  k_pack_w<<<(2 * 8 * 512 + 255) / 256, 256, 0, stream>>>(w2, 64, 64, 128, Bp2);
  k_pack_w<<<(4 * 4 * 512 + 255) / 256, 256, 0, stream>>>(w3, 128, 128, 64, Bp3);
  k_pack_wc<<<(9 * 2 * 4 * 512 + 255) / 256, 256, 0, stream>>>(wc, BpC);

  // stage 1: voxel sums + feature build
  k_point_stats<<<nb256, tpb, 0, stream>>>(points, pcyl, binv, cinv,
                                           bevsum, cylsum, N);
  k_build_mvfin<<<nb256, tpb, 0, stream>>>(points, pcyl, cyl_idxs, bev_idxs,
                                           binv, cinv, bevsum, cylsum, A1, N);

  int mblk = N / 64;   // 4 m-tiles (waves) per block

  // stage 2: MLP1  [N,16(pad32)] x [16,64]
  k_gemm_bf16<4><<<mblk, 128, 0, stream>>>(A1, 32, Bp1, h1, 64, 1);
  k_colstats<<<512, 256, 0, stream>>>(h1, N, 64, 6, st1, st1 + 64);
  k_bnrelu<<<2048, 256, 0, stream>>>(h1, nullptr, A2, 64, 0,
                                     st1, st1 + 64, g1, b1, N, 6);

  // stage 3: MLP2  [N,64] x [64,128]
  k_gemm_bf16<8><<<mblk, 128, 0, stream>>>(A2, 64, Bp2, h2, 128, 2);
  k_colstats<<<512, 256, 0, stream>>>(h2, N, 128, 7, st2, st2 + 128);
  k_bnrelu<<<4096, 256, 0, stream>>>(h2, h2, nullptr, 0, 0,
                                     st2, st2 + 128, g2, b2, N, 7);
  k_pointwise_to_A3<<<2048, 256, 0, stream>>>(h2, A3, N);

  // stage 4: cyl segment-max scatter -> dense image -> bf16
  k_scatter_cylmax<<<N, FD, 0, stream>>>(h2, cinv, ccoords, dense, N);
  k_f32_to_bf16<<<4096, 256, 0, stream>>>(dense, dbf, (size_t)P * FD);

  // stage 5: 3x3 conv via implicit-GEMM WMMA + BN-ReLU
  k_conv_wmma<<<(B * GCX * (GCY / 16)) / 4, 128, 0, stream>>>(dbf, BpC, cvo);
  k_colstats<<<512, 256, 0, stream>>>(cvo, P, 64, 6, stC, stC + 64);
  k_bnrelu<<<4096, 256, 0, stream>>>(cvo, cvo, nullptr, 0, 0,
                                     stC, stC + 64, gc, bc, P, 6);

  // stage 6: bilinear gather back to points (A3 cols 64..127)
  k_bilinear<<<N, FD, 0, stream>>>(cvo, cyl_idxs, points, A3, N);

  // stage 7: MLP3  [N,128] x [128,64] -> mvf_fea (output 1)
  k_gemm_bf16<4><<<mblk, 128, 0, stream>>>(A3, 128, Bp3, h3, 64, 4);
  k_colstats<<<512, 256, 0, stream>>>(h3, N, 64, 6, st3, st3 + 64);
  k_bnrelu<<<2048, 256, 0, stream>>>(h3, out_mvf, nullptr, 0, 0,
                                     st3, st3 + 64, g3, b3, N, 6);

  // stage 8: bev segment max (output 2) + voxel coords (output 3)
  k_bevmax<<<N, FD, 0, stream>>>(out_mvf, binv, out_bev, N);
  k_voxcoords<<<(Nb + tpb - 1) / tpb, tpb, 0, stream>>>(bcoords, out_vc, Nb);
}
